// ProductQuantize_38182259261962
// MI455X (gfx1250) — compile-verified, matrix operand order checked by reading, not run on hardware
//
#include <hip/hip_runtime.h>
#include <hip/hip_bf16.h>

// ---------------------------------------------------------------------------
// ProductQuantize for MI455X (gfx1250, wave32)
//   score(n,v) = dot(x_n, e_v) - 0.5*||e_v||^2   (argmax == argmin distance;
//   ||x||^2 is constant over v so it never affects the argmin)
// Cross term via V_WMMA_F32_16X16X32_BF16; norms & gathered outputs in fp32.
// ---------------------------------------------------------------------------

typedef __attribute__((ext_vector_type(16))) __bf16       v16bf;
typedef __attribute__((ext_vector_type(8)))  float        v8f;
typedef __attribute__((ext_vector_type(4)))  unsigned int v4u;

#define GH          8
#define VC          512
#define HS          128
#define NEMB        1024
#define NROWS       16384      // B*T
#define ROWS_PER_WG 128
#define CB_STRIDE   272        // bytes per LDS codebook row: 128 bf16 + 16B pad

static __device__ __forceinline__ unsigned pk_bf16(float a, float b) {
  unsigned short lo = __builtin_bit_cast(unsigned short, (__bf16)a);
  unsigned short hi = __builtin_bit_cast(unsigned short, (__bf16)b);
  return ((unsigned)hi << 16) | (unsigned)lo;
}

// A-matrix fragment (16x32 bf16, ISA 7.12.2): lanes 0-15 hold K{0..7,16..23},
// lanes 16-31 hold K{8..15,24..31} (half = 0 or 8). Two 16B chunks per lane.
static __device__ __forceinline__ v16bf load_a_global(const float* xrow, int ks, int half) {
  const float4* p0 = (const float4*)(xrow + ks * 32 + half);
  const float4* p1 = (const float4*)(xrow + ks * 32 + 16 + half);
  float4 f0 = p0[0], f1 = p0[1], f2 = p1[0], f3 = p1[1];
  union { v16bf v; unsigned u[8]; } r;
  r.u[0] = pk_bf16(f0.x, f0.y); r.u[1] = pk_bf16(f0.z, f0.w);
  r.u[2] = pk_bf16(f1.x, f1.y); r.u[3] = pk_bf16(f1.z, f1.w);
  r.u[4] = pk_bf16(f2.x, f2.y); r.u[5] = pk_bf16(f2.z, f2.w);
  r.u[6] = pk_bf16(f3.x, f3.y); r.u[7] = pk_bf16(f3.z, f3.w);
  return r.v;
}

// B-matrix fragment (32x16 bf16): lane n holds column N = lane&15, same K split.
static __device__ __forceinline__ v16bf load_b_lds(const unsigned char* cbrow, int ks, int half) {
  union { v16bf v; v4u q[2]; } r;
  r.q[0] = *(const v4u*)(cbrow + (ks * 32 + half) * 2);
  r.q[1] = *(const v4u*)(cbrow + (ks * 32 + 16 + half) * 2);
  return r.v;
}

// ---- kernel 1: embed fp32 -> bf16 codebook + fp32 ||e||^2 (one thread/row) ----
__global__ __launch_bounds__(256) void vq_prep_kernel(
    const float* __restrict__ embed, unsigned* __restrict__ cb, float* __restrict__ en) {
  const int row = blockIdx.x * 256 + threadIdx.x;        // 0..4095 = g*512+v
  const float4* src = (const float4*)(embed + (size_t)row * HS);
  unsigned* dst = cb + (size_t)row * (HS / 2);
  float nrm = 0.0f;
#pragma unroll
  for (int i = 0; i < HS / 4; ++i) {
    float4 f = src[i];
    nrm += f.x * f.x + f.y * f.y + f.z * f.z + f.w * f.w;
    dst[i * 2 + 0] = pk_bf16(f.x, f.y);
    dst[i * 2 + 1] = pk_bf16(f.z, f.w);
  }
  en[row] = nrm;
}

// ---- kernel 2: WMMA scores + argmin + fp32 gather -------------------------
__global__ __launch_bounds__(256) void vq_argmin_wmma_kernel(
    const float* __restrict__ x, const float* __restrict__ embed,
    const unsigned* __restrict__ cb, const float* __restrict__ enorm,
    float* __restrict__ outQ, float* __restrict__ outC) {
  __shared__ __align__(16) unsigned char s_cb[VC * CB_STRIDE];  // 139264 B
  __shared__ float s_en[VC];                                    //   2048 B

  const int g    = blockIdx.x >> 7;                 // 8 groups
  const int row0 = (blockIdx.x & 127) * ROWS_PER_WG;
  const int t    = threadIdx.x;

  // Stage this group's bf16 codebook into LDS (padded rows -> conflict-free
  // ds_load_b128 in the inner loop) and its fp32 norms.
  {
    const v4u* src = (const v4u*)cb + (size_t)g * (VC * HS * 2 / 16);
#pragma unroll
    for (int i = 0; i < 32; ++i) {                  // 8192 x 16B total
      int j = t + i * 256;
      v4u val = src[j];
      *(v4u*)(s_cb + (j >> 4) * CB_STRIDE + (j & 15) * 16) = val;
    }
    s_en[t]       = enorm[g * VC + t];
    s_en[t + 256] = enorm[g * VC + t + 256];
  }
  __syncthreads();

  const int w    = t >> 5;          // wave id: 8 waves, 16 rows each
  const int lane = t & 31;
  const int half = (lane >= 16) ? 8 : 0;
  const int nsub = lane & 15;

  // Load this wave's A fragments once (reused across all 32 N-tiles).
  const float* xrow = x + (size_t)(row0 + w * 16 + nsub) * NEMB + g * HS;
  v16bf a0 = load_a_global(xrow, 0, half);
  v16bf a1 = load_a_global(xrow, 1, half);
  v16bf a2 = load_a_global(xrow, 2, half);
  v16bf a3 = load_a_global(xrow, 3, half);

  float best[8]; int bidx[8];
#pragma unroll
  for (int r = 0; r < 8; ++r) { best[r] = -3.402823466e38f; bidx[r] = 0; }

  for (int nt = 0; nt < VC / 16; ++nt) {
    const int cl = nt * 16 + nsub;                  // this lane's cluster column
    const float ci = -0.5f * s_en[cl];
    v8f c;
#pragma unroll
    for (int r = 0; r < 8; ++r) c[r] = ci;          // C seeded with -0.5*||e||^2
    const unsigned char* cbrow = s_cb + cl * CB_STRIDE;
    c = __builtin_amdgcn_wmma_f32_16x16x32_bf16(false, a0, false, load_b_lds(cbrow, 0, half), (short)0, c, false, false);
    c = __builtin_amdgcn_wmma_f32_16x16x32_bf16(false, a1, false, load_b_lds(cbrow, 1, half), (short)0, c, false, false);
    c = __builtin_amdgcn_wmma_f32_16x16x32_bf16(false, a2, false, load_b_lds(cbrow, 2, half), (short)0, c, false, false);
    c = __builtin_amdgcn_wmma_f32_16x16x32_bf16(false, a3, false, load_b_lds(cbrow, 3, half), (short)0, c, false, false);
#pragma unroll
    for (int r = 0; r < 8; ++r)
      if (c[r] > best[r]) { best[r] = c[r]; bidx[r] = cl; }
  }

  // Argmax butterfly inside each 16-lane half (C layout: N = lane&15,
  // M = r (+8 for upper half)); ties -> smaller cluster index.
#pragma unroll
  for (int off = 1; off < 16; off <<= 1) {
#pragma unroll
    for (int r = 0; r < 8; ++r) {
      float ov = __shfl_xor(best[r], off, 32);
      int   oi = __shfl_xor(bidx[r], off, 32);
      if (ov > best[r] || (ov == best[r] && oi < bidx[r])) { best[r] = ov; bidx[r] = oi; }
    }
  }

  // Emit codes + gather the winning fp32 codebook rows (bit-exact output).
#pragma unroll
  for (int m = 0; m < 16; ++m) {
    const int r = m & 7;
    const int code = __shfl(bidx[r], (m < 8) ? 0 : 16, 32);   // wave-uniform
    const int gRow = row0 + w * 16 + m;
    if (lane == 0) outC[(size_t)gRow * GH + g] = (float)code;
    const float4* er = (const float4*)(embed + ((size_t)g * VC + code) * HS);
    float4 v = er[lane];                                       // 32 lanes x 16B = 512B row
    ((float4*)(outQ + (size_t)gRow * NEMB + g * HS))[lane] = v;
  }
}

// ---------------------------------------------------------------------------
extern "C" void kernel_launch(void* const* d_in, const int* in_sizes, int n_in,
                              void* d_out, int out_size, void* d_ws, size_t ws_size,
                              hipStream_t stream) {
  const float* x     = (const float*)d_in[0];   // (8, 2048, 1024) fp32
  const float* embed = (const float*)d_in[1];   // (8, 512, 128)  fp32

  float* outQ = (float*)d_out;                          // 16,777,216 floats
  float* outC = outQ + (size_t)NROWS * NEMB;            //    131,072 codes (as float)

  unsigned* cb = (unsigned*)d_ws;                                   // 1 MB bf16 codebook
  float*    en = (float*)((char*)d_ws + (size_t)GH * VC * HS * 2);  // 16 KB norms

  vq_prep_kernel<<<GH * VC / 256, 256, 0, stream>>>(embed, cb, en);
  vq_argmin_wmma_kernel<<<GH * (NROWS / ROWS_PER_WG), 256, 0, stream>>>(
      x, embed, cb, en, outQ, outC);
}
